// BitConv2dInfer_12910671692152
// MI455X (gfx1250) — compile-verified, matrix operand order checked by reading, not run on hardware
//
#include <hip/hip_runtime.h>
#include <stdint.h>

typedef __attribute__((ext_vector_type(16))) _Float16 v16h;
typedef __attribute__((ext_vector_type(8)))  _Float16 v8h;
typedef __attribute__((ext_vector_type(8)))  float    v8f;

#define C_IN   128
#define C_OUT  256
#define HH     56
#define WW_    56
#define HW     3136        // 56*56
#define NIMG   32
#define WDIM   68          // staged w-slots per row (w' = -1..62 staged, +4 zero pad)
#define PXS    40          // halves per pixel slot (32 cin + 8 pad) -> 80B stride,
                           // ds_load_b128 across 16 lanes is bank-conflict free
#define BUFH   (4 * WDIM * PXS)   // halves per LDS buffer (10880)

// ---------------------------------------------------------------------------
// Kernel 1: swizzle ternary int8 weights into f16 WMMA A-fragments in d_ws.
// Fragment f = (tap*4 + cin_chunk)*16 + co_tile ; 32 lanes x 16 halves, each
// lane's 32 bytes contiguous (two coalesced global_load_b128 in the main
// kernel).  Half j -> K per the ISA 16-bit A 16x32 layout:
//   lane<16:  K = j<8 ? j   : j+8      (K 0..7, 16..23)
//   lane>=16: K = j<8 ? j+8 : j+16     (K 8..15, 24..31)
// ---------------------------------------------------------------------------
__global__ void prep_weights(const int8_t* __restrict__ wq,
                             _Float16* __restrict__ wf) {
    int gid = blockIdx.x * blockDim.x + threadIdx.x;
    if (gid >= 576 * 32) return;
    int lane = gid & 31;
    int f    = gid >> 5;            // 0..575
    int cot  = f & 15;              // co tile (16 c_out each)
    int kc   = (f >> 4) & 3;        // cin chunk of 32
    int t    = f >> 6;              // tap 0..8  (kh = t/3, kw = t%3)
    int co   = cot * 16 + (lane & 15);
    int khi  = (lane >> 4) & 1;

    _Float16 h[16];
#pragma unroll
    for (int j = 0; j < 16; ++j) {
        int K   = (j < 8 ? j : j + 8) + khi * 8;
        int cin = kc * 32 + K;
        h[j] = (_Float16)(float)wq[(co * C_IN + cin) * 9 + t];
    }
    v8h lo, hi;
#pragma unroll
    for (int j = 0; j < 8; ++j) { lo[j] = h[j]; hi[j] = h[j + 8]; }
    v8h* dst = (v8h*)(wf + (size_t)f * 512 + lane * 16);
    dst[0] = lo;
    dst[1] = hi;
}

// ---- staging helpers ------------------------------------------------------
__device__ __forceinline__ void stage_load(const float* __restrict__ xrow,
                                           bool rowv, int wseg, int cg, int kc,
                                           float st[32]) {
#pragma unroll
    for (int k = 0; k < 4; ++k) {
        const int wp = wseg * 4 + k - 1;            // w' of slot
        const bool v = rowv && ((unsigned)wp < WW_);
#pragma unroll
        for (int i = 0; i < 8; ++i) {
            const int cin = kc * 32 + cg * 8 + i;
            st[k * 8 + i] = v ? xrow[(size_t)cin * HW + wp] : 0.f;
        }
    }
}

__device__ __forceinline__ void stage_store(_Float16* __restrict__ buf,
                                            int rr, int wseg, int cg,
                                            const float st[32]) {
#pragma unroll
    for (int k = 0; k < 4; ++k) {
        v8h pk;
#pragma unroll
        for (int i = 0; i < 8; ++i) pk[i] = (_Float16)st[k * 8 + i];
        *(v8h*)&buf[((rr * WDIM) + wseg * 4 + k) * PXS + cg * 8] = pk;
    }
}

// ---------------------------------------------------------------------------
// Kernel 2: implicit-GEMM conv, halo-staged, LDS double-buffered.
// Block = 256 threads = 8 waves = 4 co-waves x 2 px-waves.
// WG tile: 128 c_out x (2 rows x 56 w = 112 px, padded to 128).
// Per cin-chunk of 32: next chunk's global loads are issued BEFORE the
// 9-tap / 72-WMMA compute, converted + stored to the other LDS buffer after.
// One barrier per chunk.
// ---------------------------------------------------------------------------
__global__ __launch_bounds__(256)
void bitconv_wmma(const float* __restrict__ x,
                  const _Float16* __restrict__ wf,
                  const float* __restrict__ s,
                  const float* __restrict__ bias,
                  float* __restrict__ out) {
    __shared__ __align__(16) _Float16 ldsX[2][BUFH];   // 2 x 21760 B

    const int tid    = threadIdx.x;
    const int lane   = tid & 31;
    const int wave   = tid >> 5;
    const int waveCo = wave & 3;      // 4 waves cover 128 c_out
    const int wavePx = wave >> 2;     // row (0/1) of the 2-row pixel tile
    const int laneLo = lane & 15;
    const int hi16   = (lane >> 4) & 1;

    const int n  = blockIdx.x / 28;          // image
    const int h0 = (blockIdx.x % 28) * 2;    // first output row of tile
    const float* xn = x + (size_t)n * C_IN * HW;

    // ---- staging role: thread stages 4 w-slots x 8 cins of one halo row ----
    const int cg   = tid >> 6;        // cin group: cins 8*cg .. 8*cg+7
    const int slot = tid & 63;
    const int rr   = slot >> 4;       // staged row 0..3  (h = h0-1+rr)
    const int wseg = slot & 15;       // w-slot group of 4
    const int hr   = h0 + rr - 1;
    const bool rowv = (unsigned)hr < HH;
    const float* xrow = xn + hr * WW_;

    // zero the never-staged pad slots (widx 64..67) in BOTH buffers:
    // 8 regions x 160 contiguous halves = 160 single v8h stores
    if (tid < 160) {
        const int buf = tid / 80;
        const int zr  = (tid % 80) / 20;
        const int off = (tid % 20) * 8;
        v8h z = {};
        *(v8h*)&ldsX[buf][(zr * WDIM + 64) * PXS + off] = z;
    }

    v8f acc[2][4];
    {
        v8f z = {};
#pragma unroll
        for (int i = 0; i < 2; ++i)
#pragma unroll
            for (int j = 0; j < 4; ++j) acc[i][j] = z;
    }

    // ---- prologue: stage chunk 0 into buffer 0 ----
    float st[32];
    stage_load(xrow, rowv, wseg, cg, 0, st);
    stage_store(&ldsX[0][0], rr, wseg, cg, st);
    __syncthreads();

    for (int kc = 0; kc < 4; ++kc) {
        const _Float16* __restrict__ Xb = &ldsX[kc & 1][0];

        // issue next chunk's global loads now; latency hidden by 72 WMMAs
        if (kc < 3) stage_load(xrow, rowv, wseg, cg, kc + 1, st);

        // ---- all 9 taps from LDS: 72 WMMAs per wave ----
#pragma unroll
        for (int t = 0; t < 9; ++t) {
            const int dh = t / 3 - 1, dw = t % 3 - 1;
            // A fragments (pre-swizzled weights, L2-resident)
            v16h a[2];
#pragma unroll
            for (int i = 0; i < 2; ++i) {
                const int cot = blockIdx.y * 8 + waveCo * 2 + i;
                const int f   = (t * 4 + kc) * 16 + cot;
                a[i] = *(const v16h*)(wf + (size_t)f * 512 + lane * 16);
            }
            const int rrow = wavePx + dh + 1;     // staged row index 0..3
            // B fragments: lane's pixel = (row h0+wavePx+dh, w = j*16+laneLo+dw)
            v16h b[4];
#pragma unroll
            for (int j = 0; j < 4; ++j) {
                const int widx = j * 16 + laneLo + dw + 1;
                const v8h* bp =
                    (const v8h*)&Xb[(rrow * WDIM + widx) * PXS + hi16 * 8];
                v8h blo = bp[0];          // halves 0..7   (or 8..15)
                v8h bhi = bp[2];          // halves 16..23 (or 24..31)
                b[j] = __builtin_shufflevector(blo, bhi,
                         0,1,2,3,4,5,6,7,8,9,10,11,12,13,14,15);
            }
            // i outer: adjacent WMMAs share the A operand
#pragma unroll
            for (int i = 0; i < 2; ++i)
#pragma unroll
                for (int j = 0; j < 4; ++j)
                    acc[i][j] = __builtin_amdgcn_wmma_f32_16x16x32_f16(
                        false, a[i], false, b[j], (short)0, acc[i][j],
                        false, false);
        }

        // convert + publish next chunk into the other buffer
        if (kc < 3) {
            stage_store(&ldsX[(kc + 1) & 1][0], rr, wseg, cg, st);
            __syncthreads();
        }
    }

    // ---- epilogue: y = acc * s[co] + bias[co], fp32 NCHW stores ----
    const int coWg = blockIdx.y * 128 + waveCo * 32;
    const int hi8  = hi16 * 8;
    float sv[2][8], bv[2][8];
#pragma unroll
    for (int i = 0; i < 2; ++i)
#pragma unroll
        for (int r = 0; r < 8; ++r) {
            const int co = coWg + i * 16 + r + hi8;   // C/D 16x16 f32 layout
            sv[i][r] = s[co];
            bv[i][r] = bias[co];
        }

    const int hout = h0 + wavePx;
    float* orow = out + (size_t)n * C_OUT * HW + (size_t)hout * WW_;
#pragma unroll
    for (int j = 0; j < 4; ++j) {
        const int wout = j * 16 + laneLo;
        if (wout < WW_) {                       // mask the 8 padded columns
#pragma unroll
            for (int i = 0; i < 2; ++i)
#pragma unroll
                for (int r = 0; r < 8; ++r) {
                    const int co = coWg + i * 16 + r + hi8;
                    orow[(size_t)co * HW + wout] =
                        acc[i][j][r] * sv[i][r] + bv[i][r];
                }
        }
    }
}

// ---------------------------------------------------------------------------
extern "C" void kernel_launch(void* const* d_in, const int* in_sizes, int n_in,
                              void* d_out, int out_size, void* d_ws, size_t ws_size,
                              hipStream_t stream) {
    const float*  x   = (const float*)d_in[0];
    const int8_t* wq  = (const int8_t*)d_in[1];   // ternary int8 weights
    const float*  sc  = (const float*)d_in[2];    // per-c_out scale (256)
    const float*  bi  = (const float*)d_in[3];    // bias (256)
    float*        out = (float*)d_out;
    _Float16*     wf  = (_Float16*)d_ws;          // 576*512 halves = 576 KiB

    // 1) swizzle weights into WMMA fragment order (tiny, once per launch)
    prep_weights<<<72, 256, 0, stream>>>(wq, wf);

    // 2) conv: grid = (N * 28 row-pair tiles, 2 c_out halves)
    dim3 grid(NIMG * 28, C_OUT / 128);
    bitconv_wmma<<<grid, 256, 0, stream>>>(x, wf, sc, bi, out);
}
